// AutoCorrelation_65240553226425
// MI455X (gfx1250) — matrix operand order repaired and checked by reading, not compiled, not run on hardware
//
#include <hip/hip_runtime.h>
#include <hip/hip_bf16.h>
#include <math.h>

typedef __attribute__((ext_vector_type(16))) _Float16 v16h;
typedef __attribute__((ext_vector_type(8)))  float    v8f;

#define NB 32
#define LL 1024
#define HH 8
#define EE 64
#define TOPK 6

// -----------------------------------------------------------------------------
// Circular cross-correlation via WMMA.
// Grid: N*H*4 blocks of 64 threads (2 wave32). Each wave owns 8 e-channels,
// staged two-at-a-time with float2 loads. For one series (n,h,e):
//   A_c[i,s] = q[(tau0 + 32c + i + s) mod L]   (Hankel tile, 16x32 f16)
//   B_c[s,j] = k[(32c + 16j + s) mod L]        (shifted segments, 32x16 f16)
//   D[i,j]   = sum_c A_c B_c = corr[(tau0 + i - 16j) mod L]  -> 256 lags/tile
// B_c is independent of tau0 => c-outer loop with 4 accumulators loads B once.
// -----------------------------------------------------------------------------
__global__ __launch_bounds__(64)
void corr_wmma_kernel(const float* __restrict__ Q, const float* __restrict__ K,
                      float* __restrict__ corr_out, float* __restrict__ mean_g)
{
    __shared__ __attribute__((aligned(16))) _Float16 sQ [2][2][2048]; // wrap-duplicated q
    __shared__ __attribute__((aligned(16))) _Float16 sQs[2][2][2048]; // q shifted +1 elem
    __shared__ __attribute__((aligned(16))) _Float16 sK [2][2][1280]; // wrap-extended k
    __shared__ float mean_local[LL];

    const int tid  = threadIdx.x;
    const int w    = tid >> 5;          // wave 0..1
    const int lane = tid & 31;
    const int bid  = blockIdx.x;        // ((n*8 + h)*4 + quarter)
    const int qtr  = bid & 3;
    const int h    = (bid >> 2) & 7;
    const int n    = bid >> 5;

    for (int i = tid; i < LL; i += 64) mean_local[i] = 0.0f;
    __syncthreads();

    const int m  = lane & 15;           // A row / B column
    const int hb = lane >> 4;           // lane half selects K sub-block
    const int j  = m;

    int KA[8], KB[8];
    #pragma unroll
    for (int r = 0; r < 8; ++r) {
        KA[r] = ((r & 4) << 2) + (hb << 3) + ((r & 3) << 1); // A: {0|16}+8*hb+2r'
        KB[r] = ((r & 4) << 2) + (hb << 3) + ((r & 3) << 1); // B: {0|16}+8*hb+2r'
    }

    const size_t nh_base = (size_t)n * LL * HH * EE + (size_t)h * EE;

    for (int rr = 0; rr < 4; ++rr) {
        const int e0 = qtr * 16 + w * 8 + rr * 2;   // even => float2 loads aligned
        const float* qg = Q + nh_base + e0;
        const float* kg = K + nh_base + e0;

        _Float16* qe0 = sQ [w][0]; _Float16* qe1 = sQ [w][1];
        _Float16* qs0 = sQs[w][0]; _Float16* qs1 = sQs[w][1];
        _Float16* ke0 = sK [w][0]; _Float16* ke1 = sK [w][1];

        // Stage two adjacent channels (e0, e0+1) as f16 with circular wrap
        // plus a +1-shifted q copy (keeps all A-operand fetches 4B aligned).
        for (int l = lane; l < LL; l += 32) {
            const float2 q2 = *(const float2*)(qg + (size_t)l * (HH * EE));
            const float2 k2 = *(const float2*)(kg + (size_t)l * (HH * EE));
            const _Float16 a0 = (_Float16)q2.x, a1 = (_Float16)q2.y;
            const _Float16 b0 = (_Float16)k2.x, b1 = (_Float16)k2.y;
            const int jm = (l + 1023) & 1023;       // qs[i] = q[(i+1) mod L]
            qe0[l] = a0; qe0[l + 1024] = a0; qs0[jm] = a0; qs0[jm + 1024] = a0;
            qe1[l] = a1; qe1[l + 1024] = a1; qs1[jm] = a1; qs1[jm + 1024] = a1;
            ke0[l] = b0;
            ke1[l] = b1;
            if (l < 256) { ke0[l + 1024] = b0; ke1[l + 1024] = b1; }  // k idx max 1263
        }
        __syncthreads();

        for (int slot = 0; slot < 2; ++slot) {
            const int e = e0 + slot;
            const _Float16* qe = sQ [w][slot];
            const _Float16* qs = sQs[w][slot];
            const _Float16* ke = sK [w][slot];
            // A element parity == parity of m: odd lanes use the shifted copy.
            const char* qb = (m & 1) ? ((const char*)qs - 2) : (const char*)qe;

            v8f acc0 = {}, acc1 = {}, acc2 = {}, acc3 = {};
            for (int c = 0; c < 32; ++c) {
                union { v16h v; unsigned u[8]; } B, A0, A1, A2, A3;
                const int bbase = 32 * c + (j << 4);
                #pragma unroll
                for (int r = 0; r < 8; ++r)
                    B.u[r] = *(const unsigned*)((const char*)ke + 2 * (bbase + KB[r]));

                const int abase = 32 * c + m;
                #pragma unroll
                for (int r = 0; r < 8; ++r)
                    A0.u[r] = *(const unsigned*)(qb + 2 * (abase +  240 + KA[r]));
                acc0 = __builtin_amdgcn_wmma_f32_16x16x32_f16(
                           false, A0.v, false, B.v, (short)0, acc0, false, false);
                #pragma unroll
                for (int r = 0; r < 8; ++r)
                    A1.u[r] = *(const unsigned*)(qb + 2 * (abase +  496 + KA[r]));
                acc1 = __builtin_amdgcn_wmma_f32_16x16x32_f16(
                           false, A1.v, false, B.v, (short)0, acc1, false, false);
                #pragma unroll
                for (int r = 0; r < 8; ++r)
                    A2.u[r] = *(const unsigned*)(qb + 2 * (abase +  752 + KA[r]));
                acc2 = __builtin_amdgcn_wmma_f32_16x16x32_f16(
                           false, A2.v, false, B.v, (short)0, acc2, false, false);
                #pragma unroll
                for (int r = 0; r < 8; ++r)
                    A3.u[r] = *(const unsigned*)(qb + 2 * (abase + 1008 + KA[r]));
                acc3 = __builtin_amdgcn_wmma_f32_16x16x32_f16(
                           false, A3.v, false, B.v, (short)0, acc3, false, false);
            }

            const size_t obase = nh_base + e;       // + l*512 per element
            #pragma unroll
            for (int t = 0; t < 4; ++t) {
                const v8f acc = (t == 0) ? acc0 : (t == 1) ? acc1
                              : (t == 2) ? acc2 : acc3;
                const int tau0 = 240 + 256 * t;
                #pragma unroll
                for (int g = 0; g < 8; ++g) {
                    const int irow = g + (hb << 3);
                    const int l    = (tau0 + irow - (j << 4)) & 1023;
                    const float val = acc[g];
                    corr_out[obase + (size_t)l * (HH * EE)] = val;
                    atomicAdd(&mean_local[l], val);
                }
            }
        }
        __syncthreads();
    }

    // Flush this block's partial (h,e)-mean (16 of 512 series), scaled 1/(H*E).
    for (int i = tid; i < LL; i += 64)
        atomicAdd(&mean_g[n * LL + i], mean_local[i] * (1.0f / (HH * EE)));
}

// -----------------------------------------------------------------------------
// Global mean over n, serial top-6, per-n softmax weights. One block.
// -----------------------------------------------------------------------------
__global__ __launch_bounds__(256)
void stats_kernel(const float* __restrict__ mean_g, int* __restrict__ topk,
                  float* __restrict__ tmpcorr)
{
    __shared__ float g[LL];
    __shared__ int   sidx[TOPK];
    const int tid = threadIdx.x;

    for (int t = tid; t < LL; t += 256) {
        float s = 0.0f;
        for (int nn = 0; nn < NB; ++nn) s += mean_g[nn * LL + t];
        g[t] = s * (1.0f / NB);
    }
    __syncthreads();

    if (tid == 0) {
        for (int k = 0; k < TOPK; ++k) {
            int best = 0; float bv = g[0];
            for (int t = 1; t < LL; ++t)
                if (g[t] > bv) { bv = g[t]; best = t; }
            sidx[k] = best;
            topk[k] = best;
            g[best] = -3.0e38f;
        }
    }
    __syncthreads();

    if (tid < NB) {
        const int nn = tid;
        float wv[TOPK], mx = -3.0e38f;
        for (int k = 0; k < TOPK; ++k) {
            wv[k] = mean_g[nn * LL + sidx[k]];
            mx = fmaxf(mx, wv[k]);
        }
        float sum = 0.0f;
        for (int k = 0; k < TOPK; ++k) { wv[k] = __expf(wv[k] - mx); sum += wv[k]; }
        const float inv = 1.0f / sum;
        for (int k = 0; k < TOPK; ++k) tmpcorr[nn * TOPK + k] = wv[k] * inv;
    }
}

// -----------------------------------------------------------------------------
// out[n,l,h,e] = sum_k tmpcorr[n,k] * v[n, (l+idx_k) mod L, h, e]
// Fully coalesced (consecutive threads -> consecutive e). Memory-bound.
// -----------------------------------------------------------------------------
__global__ __launch_bounds__(256)
void agg_kernel(const float* __restrict__ V, const int* __restrict__ topk,
                const float* __restrict__ tmpcorr, float* __restrict__ out)
{
    const size_t t = (size_t)blockIdx.x * 256 + threadIdx.x;  // over N*L*H*E
    const int e = (int)(t & 63);
    const int h = (int)((t >> 6) & 7);
    const int l = (int)((t >> 9) & 1023);
    const int n = (int)(t >> 19);

    float acc = 0.0f;
    #pragma unroll
    for (int k = 0; k < TOPK; ++k) {
        const int lk = (l + topk[k]) & 1023;
        acc += tmpcorr[n * TOPK + k] *
               V[(((size_t)n * LL + lk) * HH + h) * EE + e];
    }
    out[t] = acc;
}

// -----------------------------------------------------------------------------
extern "C" void kernel_launch(void* const* d_in, const int* in_sizes, int n_in,
                              void* d_out, int out_size, void* d_ws, size_t ws_size,
                              hipStream_t stream)
{
    const float* Q = (const float*)d_in[0];
    const float* K = (const float*)d_in[1];
    const float* V = (const float*)d_in[2];

    const size_t elems = (size_t)NB * LL * HH * EE;   // 16,777,216
    float* out      = (float*)d_out;
    float* corr_out = out + elems;

    float* mean_g  = (float*)d_ws;                                     // N*L floats
    int*   topk    = (int*)((char*)d_ws + (size_t)NB * LL * 4);        // 6 ints (pad 64B)
    float* tmpcorr = (float*)((char*)d_ws + (size_t)NB * LL * 4 + 64); // N*6 floats

    hipMemsetAsync(d_ws, 0, (size_t)NB * LL * sizeof(float), stream);

    corr_wmma_kernel<<<NB * HH * 4, 64, 0, stream>>>(Q, K, corr_out, mean_g);
    stats_kernel<<<1, 256, 0, stream>>>(mean_g, topk, tmpcorr);
    agg_kernel<<<(int)(elems / 256), 256, 0, stream>>>(V, topk, tmpcorr, out);
}
